// LocalMemoryDecoder_30966714204736
// MI455X (gfx1250) — matrix operand order repaired and checked by reading, not compiled, not run on hardware
//
#include <hip/hip_runtime.h>
#include <hip/hip_bf16.h>
#include <math.h>

// Problem constants (from reference)
#define Vv 20000
#define Dd 512
#define Bb 32
#define Tt 32
#define Ss 200
#define Rr 100
#define Cc 8
#define SOS_TOK 2

typedef __attribute__((ext_vector_type(16))) __bf16 v16bf;
typedef __attribute__((ext_vector_type(8)))  __bf16 v8bf;
typedef __attribute__((ext_vector_type(8)))  float  v8f;

// ---------------------------------------------------------------------------
// f32 -> bf16 (RNE) elementwise conversion
// ---------------------------------------------------------------------------
__global__ void cvt_f32_bf16_kernel(const float* __restrict__ in,
                                    unsigned short* __restrict__ out, int n) {
    int i = blockIdx.x * blockDim.x + threadIdx.x;
    if (i < n) {
        unsigned u = __float_as_uint(in[i]);
        unsigned r = (u + 0x7FFFu + ((u >> 16) & 1u)) >> 16;   // round-nearest-even
        out[i] = (unsigned short)r;
    }
}

// ---------------------------------------------------------------------------
// W_hh [3D, D] -> Whh_t [D, 3D]  (tiled transpose, coalesced both sides)
// grid (D/32, 3D/32), block (32, 8)
// ---------------------------------------------------------------------------
__global__ void transpose_whh_kernel(const float* __restrict__ in,   // [3D, D]
                                     float* __restrict__ out) {      // [D, 3D]
    __shared__ float tile[32][33];
    int gx = blockIdx.x * 32;   // k (input col)
    int gy = blockIdx.y * 32;   // g (input row)
    int x = gx + threadIdx.x;
#pragma unroll
    for (int j = 0; j < 4; ++j) {
        int y = gy + threadIdx.y + j * 8;
        tile[threadIdx.y + j * 8][threadIdx.x] = in[(size_t)y * Dd + x];
    }
    __syncthreads();
    int xo = gy + threadIdx.x;  // g (output col)
#pragma unroll
    for (int j = 0; j < 4; ++j) {
        int yo = gx + threadIdx.y + j * 8;  // k (output row)
        out[(size_t)yo * (3 * Dd) + xo] = tile[threadIdx.x][threadIdx.y + j * 8];
    }
}

// ---------------------------------------------------------------------------
// h0 = relu(encode_hidden @ W_proj^T + b_proj)   [B, D]   (tiny, one-shot)
// ---------------------------------------------------------------------------
__global__ void h0_kernel(const float* __restrict__ eh,     // [B, 2D]
                          const float* __restrict__ Wp,     // [D, 2D]
                          const float* __restrict__ bp,     // [D]
                          float* __restrict__ h0) {         // [B, D]
    int idx = blockIdx.x * blockDim.x + threadIdx.x;        // B*D threads
    int b = idx / Dd, d = idx % Dd;
    const float* er = eh + (size_t)b * (2 * Dd);
    const float* wr = Wp + (size_t)d * (2 * Dd);
    float acc = bp[d];
    for (int k = 0; k < 2 * Dd; ++k) acc += er[k] * wr[k];
    h0[idx] = acc > 0.f ? acc : 0.f;
}

// ---------------------------------------------------------------------------
// x_gates via WMMA: A rows = C_bf16[tok(m)] (gathered), B = W_ih_bf16 (NT GEMM)
// x_gates[m, g] = emb[m] . W_ih[g] + b_ih[g],  m = t*B+b, M=1024, N=1536, K=512
// One wave/block, 4 M-tiles/wave.  grid (N/16 = 96, M/64 = 16), block 32.
// ---------------------------------------------------------------------------
__global__ __launch_bounds__(32)
void xgates_wmma_kernel(const unsigned short* __restrict__ Cbf,   // [V, D] bf16
                        const unsigned short* __restrict__ Wbf,   // [3D, D] bf16
                        const float* __restrict__ bih,            // [3D]
                        const int*   __restrict__ tgt,            // [B, T]
                        float* __restrict__ xg) {                 // [T*B, 3D]
    const __bf16* Cb = reinterpret_cast<const __bf16*>(Cbf);
    const __bf16* Wb = reinterpret_cast<const __bf16*>(Wbf);
    const int lane  = threadIdx.x & 31;
    const int half  = lane >> 4;
    const int l16   = lane & 15;
    const int g     = blockIdx.x * 16 + l16;    // gate column
    const int mbase = blockIdx.y * 64;

    const __bf16* bptr = Wb + (size_t)g * Dd + half * 8;
    // Per-M-tile gathered A row pointers (token fixed per row)
    const __bf16* aptr[4];
#pragma unroll
    for (int mt = 0; mt < 4; ++mt) {
        int m = mbase + mt * 16 + l16;
        int t = m >> 5, b = m & 31;
        int tok = (t == 0) ? SOS_TOK : tgt[b * Tt + (t - 1)];
        aptr[mt] = Cb + (size_t)tok * Dd + half * 8;
    }
    v8f acc[4] = {};
    for (int k0 = 0; k0 < Dd; k0 += 32) {
        v16bf bfrag = __builtin_shufflevector(
            *(const v8bf*)(bptr + k0), *(const v8bf*)(bptr + k0 + 16),
            0, 1, 2, 3, 4, 5, 6, 7, 8, 9, 10, 11, 12, 13, 14, 15);
#pragma unroll
        for (int mt = 0; mt < 4; ++mt) {
            v8bf lo = *(const v8bf*)(aptr[mt] + k0);
            v8bf hi = *(const v8bf*)(aptr[mt] + k0 + 16);
            v16bf afrag = __builtin_shufflevector(
                lo, hi, 0, 1, 2, 3, 4, 5, 6, 7, 8, 9, 10, 11, 12, 13, 14, 15);
            acc[mt] = __builtin_amdgcn_wmma_f32_16x16x32_bf16(
                false, afrag, false, bfrag, (short)0, acc[mt], false, false);
        }
    }
    float bias = bih[g];
#pragma unroll
    for (int mt = 0; mt < 4; ++mt) {
#pragma unroll
        for (int e = 0; e < 8; ++e) {
            int row = mbase + mt * 16 + half * 8 + e;
            xg[(size_t)row * (3 * Dd) + g] = acc[mt][e] + bias;
        }
    }
}

// ---------------------------------------------------------------------------
// One GRU step (exact f32). Whh_t is [D, 3D] so the k-loop loads are coalesced.
// grid 64 x 256.
// ---------------------------------------------------------------------------
__global__ void gru_step_kernel(const float* __restrict__ h_prev, // [B, D]
                                const float* __restrict__ xg_t,   // [B, 3D]
                                const float* __restrict__ Whh_t,  // [D, 3D]
                                const float* __restrict__ bhh,    // [3D]
                                float* __restrict__ h_out) {      // [B, D]
    __shared__ float s_h[Dd];
    int tid = threadIdx.x;
    int b   = blockIdx.x >> 1;
    int d   = ((blockIdx.x & 1) << 8) + tid;
    s_h[tid]       = h_prev[b * Dd + tid];
    s_h[tid + 256] = h_prev[b * Dd + tid + 256];
    __syncthreads();
    float hr = bhh[d], hz = bhh[Dd + d], hn = bhh[2 * Dd + d];
    for (int k = 0; k < Dd; ++k) {
        float h = s_h[k];
        const float* w = Whh_t + (size_t)k * (3 * Dd);
        hr += h * w[d];
        hz += h * w[Dd + d];
        hn += h * w[2 * Dd + d];
    }
    float xr = xg_t[b * 3 * Dd + d];
    float xz = xg_t[b * 3 * Dd + Dd + d];
    float xn = xg_t[b * 3 * Dd + 2 * Dd + d];
    float r = 1.f / (1.f + expf(-(xr + hr)));
    float z = 1.f / (1.f + expf(-(xz + hz)));
    float n = tanhf(xn + r * hn);
    h_out[b * Dd + d] = (1.f - z) * n + z * s_h[d];
}

// ---------------------------------------------------------------------------
// p_vocab = hs_bf16 [1024,512] x C_bf16^T (NT GEMM).  One wave/block,
// 8 M-tiles/wave (B fragment reused 8x -> halves C_bf16 L2 traffic).
// grid (V/16 = 1250, M/128 = 8), block 32.
// ---------------------------------------------------------------------------
__global__ __launch_bounds__(32)
void pvocab_wmma_kernel(const unsigned short* __restrict__ Aus,  // hs bf16 [1024,512]
                        const unsigned short* __restrict__ Bus,  // C  bf16 [20000,512]
                        float* __restrict__ out) {               // [1024, 20000]
    const __bf16* A  = reinterpret_cast<const __bf16*>(Aus);
    const __bf16* Bm = reinterpret_cast<const __bf16*>(Bus);
    const int lane  = threadIdx.x & 31;
    const int half  = lane >> 4;
    const int l16   = lane & 15;
    const int ncol  = blockIdx.x * 16 + l16;
    const int mbase = blockIdx.y * 128;

    const __bf16* bptr  = Bm + (size_t)ncol * Dd + half * 8;
    const __bf16* aptr0 = A + (size_t)(mbase + l16) * Dd + half * 8;

    v8f acc[8] = {};
    for (int k0 = 0; k0 < Dd; k0 += 32) {
        v16bf bfrag = __builtin_shufflevector(
            *(const v8bf*)(bptr + k0), *(const v8bf*)(bptr + k0 + 16),
            0, 1, 2, 3, 4, 5, 6, 7, 8, 9, 10, 11, 12, 13, 14, 15);
#pragma unroll
        for (int mt = 0; mt < 8; ++mt) {
            const __bf16* ap = aptr0 + (size_t)mt * 16 * Dd + k0;
            v8bf lo = *(const v8bf*)(ap);        // K = k0+half*8 .. +8
            v8bf hi = *(const v8bf*)(ap + 16);   // K = k0+16+half*8 .. +8
            v16bf afrag = __builtin_shufflevector(
                lo, hi, 0, 1, 2, 3, 4, 5, 6, 7, 8, 9, 10, 11, 12, 13, 14, 15);
            acc[mt] = __builtin_amdgcn_wmma_f32_16x16x32_bf16(
                false, afrag, false, bfrag, (short)0, acc[mt], false, false);
        }
    }
#pragma unroll
    for (int mt = 0; mt < 8; ++mt) {
#pragma unroll
        for (int e = 0; e < 8; ++e) {
            int row = mbase + mt * 16 + half * 8 + e;
            out[(size_t)row * Vv + (size_t)blockIdx.x * 16 + l16] = acc[mt][e];
        }
    }
}

// ---------------------------------------------------------------------------
// story_logits[t,b,n] = (hs[t,b,:] . story_mem[b,n,:]) * story_ptr[b,n]
// grid = T*B blocks, 256 threads
// ---------------------------------------------------------------------------
__global__ void story_kernel(const float* __restrict__ hs,   // [T,B,D]
                             const float* __restrict__ mem,  // [B,S,D]
                             const float* __restrict__ gp,   // [B,S]
                             float* __restrict__ out) {      // [T,B,S]
    __shared__ float s_h[Dd];
    int tb = blockIdx.x, tid = threadIdx.x;
    int b = tb % Bb;
    s_h[tid]       = hs[(size_t)tb * Dd + tid];
    s_h[tid + 256] = hs[(size_t)tb * Dd + tid + 256];
    __syncthreads();
    if (tid < Ss) {
        const float* mr = mem + ((size_t)b * Ss + tid) * Dd;
        float acc = 0.f;
        for (int k = 0; k < Dd; ++k) acc += s_h[k] * mr[k];
        out[(size_t)tb * Ss + tid] = acc * gp[b * Ss + tid];
    }
}

// kb_logits[t,b,r] = (hs . kb_mem[b,r,:]) * kb_ptr[b,r]   grid T*B, 128 thr
__global__ void kb_kernel(const float* __restrict__ hs,
                          const float* __restrict__ mem,   // [B,R,D]
                          const float* __restrict__ gp,    // [B,R]
                          float* __restrict__ out) {       // [T,B,R]
    __shared__ float s_h[Dd];
    int tb = blockIdx.x, tid = threadIdx.x;
    int b = tb % Bb;
    for (int j = 0; j < 4; ++j) s_h[tid + j * 128] = hs[(size_t)tb * Dd + tid + j * 128];
    __syncthreads();
    if (tid < Rr) {
        const float* mr = mem + ((size_t)b * Rr + tid) * Dd;
        float acc = 0.f;
        for (int k = 0; k < Dd; ++k) acc += s_h[k] * mr[k];
        out[(size_t)tb * Rr + tid] = acc * gp[b * Rr + tid];
    }
}

// ---------------------------------------------------------------------------
// ptr_kb: wave-cooperative column attention.  One wave per (t,b,r):
// lanes split K (coalesced b128 loads), butterfly-reduce 8 dots, softmax,
// pointer blend + length mask; lanes 0..7 store the 8 columns.
// total waves = T*B*R = 102400; block = 256 (8 waves) -> grid 12800
// ---------------------------------------------------------------------------
__global__ void ptrkb_kernel(const float* __restrict__ hs,      // [T,B,D]
                             const float* __restrict__ Cw,      // [V,D] f32
                             const int*   __restrict__ colarr,  // [B,R,C]
                             const float* __restrict__ kbl,     // [T,B,R]
                             const int*   __restrict__ klen,    // [B]
                             float* __restrict__ out) {         // [T,B,R*C]
    int wave = (blockIdx.x * blockDim.x + threadIdx.x) >> 5;    // 0..T*B*R-1
    int lane = threadIdx.x & 31;
    int r  = wave % Rr;
    int tb = wave / Rr;
    int b  = tb % Bb;

    // hs row, K split across lanes: lane covers k = 4*lane + 128*j, j=0..3
    const float* hrow = hs + (size_t)tb * Dd;
    float4 hreg[4];
#pragma unroll
    for (int j = 0; j < 4; ++j)
        hreg[j] = *(const float4*)(hrow + 4 * lane + 128 * j);

    float sc[Cc];
#pragma unroll
    for (int c = 0; c < Cc; ++c) {
        int tok = colarr[((size_t)b * Rr + r) * Cc + c];
        const float* er = Cw + (size_t)tok * Dd;
        float a = 0.f;
#pragma unroll
        for (int j = 0; j < 4; ++j) {
            float4 e4 = *(const float4*)(er + 4 * lane + 128 * j);
            a += hreg[j].x * e4.x + hreg[j].y * e4.y +
                 hreg[j].z * e4.z + hreg[j].w * e4.w;
        }
        // full butterfly: every lane ends with the complete dot product
        for (int off = 16; off > 0; off >>= 1) a += __shfl_xor(a, off, 32);
        sc[c] = a;
    }
    float mx = sc[0];
#pragma unroll
    for (int c = 1; c < Cc; ++c) mx = sc[c] > mx ? sc[c] : mx;
    float sum = 0.f;
#pragma unroll
    for (int c = 0; c < Cc; ++c) { sc[c] = expf(sc[c] - mx); sum += sc[c]; }
    float inv = 1.f / sum;
    float kl = kbl[(size_t)tb * Rr + r];
    int lastr = klen[b] - 1;
    if (lane < Cc) {
        int c = lane;
        float v = kl * (1.f - 1.f / (float)Cc) + kl * (sc[c] * inv);
        if (r == lastr && c >= 1) v = 0.f;
        out[(size_t)tb * (Rr * Cc) + r * Cc + c] = v;
    }
}

// ---------------------------------------------------------------------------
extern "C" void kernel_launch(void* const* d_in, const int* in_sizes, int n_in,
                              void* d_out, int out_size, void* d_ws, size_t ws_size,
                              hipStream_t stream) {
    (void)in_sizes; (void)n_in; (void)out_size; (void)ws_size;

    const float* Cw   = (const float*)d_in[0];
    const float* Wp   = (const float*)d_in[1];
    const float* bp   = (const float*)d_in[2];
    const float* Wih  = (const float*)d_in[3];
    const float* Whh  = (const float*)d_in[4];
    const float* bih  = (const float*)d_in[5];
    const float* bhh  = (const float*)d_in[6];
    const float* eh   = (const float*)d_in[7];
    const float* smem = (const float*)d_in[8];
    const float* kmem = (const float*)d_in[9];
    const float* sgp  = (const float*)d_in[10];
    const float* kgp  = (const float*)d_in[11];
    const int*   carr = (const int*)d_in[12];
    const int*   tgt  = (const int*)d_in[13];
    const int*   klen = (const int*)d_in[14];

    float* out = (float*)d_out;
    float* out_pv    = out;                                   // [T,B,V]
    float* out_story = out + (size_t)Tt * Bb * Vv;            // [T,B,S]
    float* out_ptrkb = out_story + (size_t)Tt * Bb * Ss;      // [T,B,R*C]

    // Workspace carve-up (256B-aligned)
    size_t off = 0;
    auto carve = [&](size_t bytes) { size_t o = off; off += (bytes + 255) & ~(size_t)255; return o; };
    char* ws = (char*)d_ws;
    float* h0      = (float*)(ws + carve(sizeof(float) * Bb * Dd));
    float* hs      = (float*)(ws + carve(sizeof(float) * Tt * Bb * Dd));
    float* xg      = (float*)(ws + carve(sizeof(float) * Tt * Bb * 3 * Dd));
    float* kbl     = (float*)(ws + carve(sizeof(float) * Tt * Bb * Rr));
    float* Whh_t   = (float*)(ws + carve(sizeof(float) * 3 * Dd * Dd));
    unsigned short* hs_bf  = (unsigned short*)(ws + carve(sizeof(unsigned short) * Tt * Bb * Dd));
    unsigned short* C_bf   = (unsigned short*)(ws + carve(sizeof(unsigned short) * (size_t)Vv * Dd));
    unsigned short* Wih_bf = (unsigned short*)(ws + carve(sizeof(unsigned short) * 3 * Dd * Dd));

    // 1) one-shot weight preps
    cvt_f32_bf16_kernel<<<(Vv * Dd) / 256, 256, 0, stream>>>(Cw, C_bf, Vv * Dd);
    cvt_f32_bf16_kernel<<<(3 * Dd * Dd) / 256, 256, 0, stream>>>(Wih, Wih_bf, 3 * Dd * Dd);
    {
        dim3 grid(Dd / 32, (3 * Dd) / 32);
        transpose_whh_kernel<<<grid, dim3(32, 8), 0, stream>>>(Whh, Whh_t);
    }
    // 2) h0
    h0_kernel<<<(Bb * Dd) / 256, 256, 0, stream>>>(eh, Wp, bp, h0);
    // 3) x_gates via WMMA (gathered-A NT GEMM, 1024 x 1536 x 512)
    {
        dim3 grid((3 * Dd) / 16, (Tt * Bb) / 64);
        xgates_wmma_kernel<<<grid, 32, 0, stream>>>(C_bf, Wih_bf, bih, tgt, xg);
    }
    // 4) GRU recurrence: 32 serial step launches (exact f32, coalesced W_hh^T)
    for (int t = 0; t < Tt; ++t) {
        const float* hp = (t == 0) ? h0 : (hs + (size_t)(t - 1) * Bb * Dd);
        gru_step_kernel<<<Bb * 2, 256, 0, stream>>>(
            hp, xg + (size_t)t * Bb * 3 * Dd, Whh_t, bhh, hs + (size_t)t * Bb * Dd);
    }
    // 5) hs -> bf16
    cvt_f32_bf16_kernel<<<(Tt * Bb * Dd) / 256, 256, 0, stream>>>(hs, hs_bf, Tt * Bb * Dd);
    // 6) p_vocab via bf16 WMMA (dominant GEMM: 1024 x 20000 x 512)
    {
        dim3 grid(Vv / 16, (Tt * Bb) / 128);
        pvocab_wmma_kernel<<<grid, 32, 0, stream>>>(hs_bf, C_bf, out_pv);
    }
    // 7) story logits
    story_kernel<<<Tt * Bb, 256, 0, stream>>>(hs, smem, sgp, out_story);
    // 8) kb logits (scratch; feeds ptr_kb)
    kb_kernel<<<Tt * Bb, 128, 0, stream>>>(hs, kmem, kgp, kbl);
    // 9) wave-cooperative column attention + pointer blend + mask
    ptrkb_kernel<<<(Tt * Bb * Rr) / 8, 256, 0, stream>>>(hs, Cw, carr, kbl, klen, out_ptrkb);
}
